// MPGDecoder_1047972020584
// MI455X (gfx1250) — matrix operand order repaired and checked by reading, not compiled, not run on hardware
//
#include <hip/hip_runtime.h>
#include <math.h>

#define DD 512
#define BB 64
#define TT 50
#define SS 400
#define KK 50
#define NEGV (-1e6f)

typedef __attribute__((ext_vector_type(16))) __bf16 v16bf;
typedef __attribute__((ext_vector_type(8)))  float  v8f;
typedef __attribute__((ext_vector_type(4)))  unsigned int u32x4;
typedef __attribute__((ext_vector_type(4)))  float  f32x4;

union FragBF {
  v16bf v;
  u32x4 u[2];
  unsigned short s[16];
};

__device__ __forceinline__ unsigned short f2bf(float f) {
  unsigned int u = __float_as_uint(f);
  unsigned int r = u + 0x7FFFu + ((u >> 16) & 1u);   // round-to-nearest-even
  return (unsigned short)(r >> 16);
}

__device__ __forceinline__ float sigf(float x) { return 1.f / (1.f + __expf(-x)); }

// ---------------------------------------------------------------------------
// Generic bf16 GEMM: C[M,N] = A[M,K] * W[N,K]^T (+bias). One wave per 16x16 tile.
// grid = (N/16, M/16), block = 32.
// ---------------------------------------------------------------------------
__global__ void gemm_bf16_kernel(const unsigned short* __restrict__ A, int lda,
                                 const unsigned short* __restrict__ W, int ldw,
                                 float* __restrict__ C, int ldc, int Kdim,
                                 const float* __restrict__ bias)
{
  const int col0 = blockIdx.x * 16;
  const int row0 = blockIdx.y * 16;
  const int lane = threadIdx.x & 31;
  const int m  = lane & 15;
  const int hi = lane >> 4;
  const unsigned short* Arow = A + (size_t)(row0 + m) * lda;
  const unsigned short* Brow = W + (size_t)(col0 + m) * ldw;
  v8f acc = {};
  for (int k0 = 0; k0 < Kdim; k0 += 32) {
    FragBF a, b;
    a.u[0] = *(const u32x4*)(Arow + k0 + hi * 8);
    a.u[1] = *(const u32x4*)(Arow + k0 + 16 + hi * 8);
    b.u[0] = *(const u32x4*)(Brow + k0 + hi * 8);
    b.u[1] = *(const u32x4*)(Brow + k0 + 16 + hi * 8);
    acc = __builtin_amdgcn_wmma_f32_16x16x32_bf16(false, a.v, false, b.v,
                                                  (short)0, acc, false, false);
  }
  const float bv = bias ? bias[col0 + m] : 0.f;
#pragma unroll
  for (int r = 0; r < 8; ++r)
    C[(size_t)(row0 + hi * 8 + r) * ldc + (col0 + m)] = acc[r] + bv;
}

// ---------------------------------------------------------------------------
// pre = ctx @ W^T + bias, reading f32 context [L, B, D] (row b*L+l), converting
// to bf16 in-register. C is [B*L, D] f32.  grid = (D/16, B*L/16), block = 32.
// ---------------------------------------------------------------------------
__global__ void gemm_pre_kernel(const float* __restrict__ ctx, int L,
                                const unsigned short* __restrict__ W,
                                const float* __restrict__ bias,
                                float* __restrict__ C)
{
  const int col0 = blockIdx.x * 16;
  const int row0 = blockIdx.y * 16;
  const int lane = threadIdx.x & 31;
  const int m  = lane & 15;
  const int hi = lane >> 4;
  const int row = row0 + m;
  const int b = row / L;
  const int l = row - b * L;
  const float* Arow = ctx + ((size_t)l * BB + b) * DD;
  const unsigned short* Brow = W + (size_t)(col0 + m) * DD;
  v8f acc = {};
  for (int k0 = 0; k0 < DD; k0 += 32) {
    FragBF a, bfr;
    f32x4 f0 = *(const f32x4*)(Arow + k0 + hi * 8);
    f32x4 f1 = *(const f32x4*)(Arow + k0 + hi * 8 + 4);
    f32x4 f2 = *(const f32x4*)(Arow + k0 + 16 + hi * 8);
    f32x4 f3 = *(const f32x4*)(Arow + k0 + 16 + hi * 8 + 4);
    a.s[0]  = f2bf(f0.x); a.s[1]  = f2bf(f0.y); a.s[2]  = f2bf(f0.z); a.s[3]  = f2bf(f0.w);
    a.s[4]  = f2bf(f1.x); a.s[5]  = f2bf(f1.y); a.s[6]  = f2bf(f1.z); a.s[7]  = f2bf(f1.w);
    a.s[8]  = f2bf(f2.x); a.s[9]  = f2bf(f2.y); a.s[10] = f2bf(f2.z); a.s[11] = f2bf(f2.w);
    a.s[12] = f2bf(f3.x); a.s[13] = f2bf(f3.y); a.s[14] = f2bf(f3.z); a.s[15] = f2bf(f3.w);
    bfr.u[0] = *(const u32x4*)(Brow + k0 + hi * 8);
    bfr.u[1] = *(const u32x4*)(Brow + k0 + 16 + hi * 8);
    acc = __builtin_amdgcn_wmma_f32_16x16x32_bf16(false, a.v, false, bfr.v,
                                                  (short)0, acc, false, false);
  }
  const float bv = bias[col0 + m];
#pragma unroll
  for (int r = 0; r < 8; ++r)
    C[((size_t)row0 + hi * 8 + r) * DD + (col0 + m)] = acc[r] + bv;
}

// ---------------------------------------------------------------------------
// Fused GRU cell: one wave owns one 16x16 tile of h_new and accumulates the
// six gate tiles (i_r,i_z,i_n,h_r,h_z,h_n) with WMMA, then combines.
// A for gi is the (possibly concatenated) input [X0 | X1], each segment width DD.
// grid = (D/16, B/16), block = 32.
// ---------------------------------------------------------------------------
__global__ void gru_step_kernel(const unsigned short* __restrict__ X0,
                                const unsigned short* __restrict__ X1,
                                int Kx,
                                const unsigned short* __restrict__ Wih,
                                const unsigned short* __restrict__ Whh,
                                const unsigned short* __restrict__ Hprev_bf,
                                const float* __restrict__ Hprev_f,
                                const float* __restrict__ bih,
                                const float* __restrict__ bhh,
                                float* __restrict__ Hnew_f,
                                unsigned short* __restrict__ Hnew_bf)
{
  const int col0 = blockIdx.x * 16;
  const int row0 = blockIdx.y * 16;
  const int lane = threadIdx.x & 31;
  const int m  = lane & 15;
  const int hi = lane >> 4;

  v8f gi[3] = {};
  v8f gh[3] = {};

  // gi = x @ Wih^T   (x = [X0 | X1])
  for (int k0 = 0; k0 < Kx; k0 += 32) {
    const unsigned short* seg = (k0 < DD) ? X0 : X1;
    const int koff = (k0 < DD) ? k0 : (k0 - DD);
    const unsigned short* Arow = seg + (size_t)(row0 + m) * DD + koff;
    FragBF a;
    a.u[0] = *(const u32x4*)(Arow + hi * 8);
    a.u[1] = *(const u32x4*)(Arow + 16 + hi * 8);
#pragma unroll
    for (int g = 0; g < 3; ++g) {
      const unsigned short* Brow = Wih + (size_t)(g * DD + col0 + m) * Kx + k0;
      FragBF b;
      b.u[0] = *(const u32x4*)(Brow + hi * 8);
      b.u[1] = *(const u32x4*)(Brow + 16 + hi * 8);
      gi[g] = __builtin_amdgcn_wmma_f32_16x16x32_bf16(false, a.v, false, b.v,
                                                      (short)0, gi[g], false, false);
    }
  }
  // gh = h_prev @ Whh^T
  for (int k0 = 0; k0 < DD; k0 += 32) {
    const unsigned short* Arow = Hprev_bf + (size_t)(row0 + m) * DD + k0;
    FragBF a;
    a.u[0] = *(const u32x4*)(Arow + hi * 8);
    a.u[1] = *(const u32x4*)(Arow + 16 + hi * 8);
#pragma unroll
    for (int g = 0; g < 3; ++g) {
      const unsigned short* Brow = Whh + (size_t)(g * DD + col0 + m) * DD + k0;
      FragBF b;
      b.u[0] = *(const u32x4*)(Brow + hi * 8);
      b.u[1] = *(const u32x4*)(Brow + 16 + hi * 8);
      gh[g] = __builtin_amdgcn_wmma_f32_16x16x32_bf16(false, a.v, false, b.v,
                                                      (short)0, gh[g], false, false);
    }
  }

  const int col = col0 + m;
#pragma unroll
  for (int r = 0; r < 8; ++r) {
    const int row = row0 + hi * 8 + r;
    float ir  = gi[0][r] + bih[col];
    float iz  = gi[1][r] + bih[DD + col];
    float in_ = gi[2][r] + bih[2 * DD + col];
    float hr  = gh[0][r] + bhh[col];
    float hz  = gh[1][r] + bhh[DD + col];
    float hn  = gh[2][r] + bhh[2 * DD + col];
    float rg = sigf(ir + hr);
    float zg = sigf(iz + hz);
    float ng = tanhf(in_ + rg * hn);
    float hp = Hprev_f[(size_t)row * DD + col];
    float hv = (1.f - zg) * ng + zg * hp;
    Hnew_f [(size_t)row * DD + col] = hv;
    Hnew_bf[(size_t)row * DD + col] = f2bf(hv);
  }
}

// ---------------------------------------------------------------------------
// gate = sigmoid(out . gW + gb).  One wave per batch row.
// ---------------------------------------------------------------------------
__global__ void gate_kernel(const float* __restrict__ h, const float* __restrict__ gW,
                            const float* __restrict__ gb,
                            float* __restrict__ gate_f, float* __restrict__ gate_out)
{
  const int b = blockIdx.x;
  const int lane = threadIdx.x;
  float s = 0.f;
  for (int d = lane; d < DD; d += 32) s += h[(size_t)b * DD + d] * gW[d];
  for (int off = 16; off > 0; off >>= 1) s += __shfl_down(s, off, 32);
  if (lane == 0) {
    float g = sigf(s + gb[0]);
    gate_f[b]   = g;
    gate_out[b] = g;
  }
}

// ---------------------------------------------------------------------------
// ConcatAttention for one batch element per block: energies, masked softmax,
// context.  ctx layout is the original [L, B, D].
// ---------------------------------------------------------------------------
__global__ void attention_kernel(const float* __restrict__ pre,
                                 const float* __restrict__ q,
                                 const float* __restrict__ vvec,
                                 const float* __restrict__ ctx,
                                 const float* __restrict__ mask,
                                 int L,
                                 float* __restrict__ c_out,
                                 float* __restrict__ a_out)
{
  const int b = blockIdx.x;
  __shared__ float qs[DD];
  __shared__ float ebuf[SS];
  __shared__ float red[8];
  const int tid  = threadIdx.x;
  const int wid  = tid >> 5;
  const int lane = tid & 31;

  for (int d = tid; d < DD; d += 256) qs[d] = q[(size_t)b * DD + d];
  __syncthreads();

  for (int l = wid; l < L; l += 8) {
    const float* pr = pre + ((size_t)b * L + l) * DD;
    float s = 0.f;
    for (int d = lane; d < DD; d += 32) s += vvec[d] * tanhf(pr[d] + qs[d]);
    for (int off = 16; off > 0; off >>= 1) s += __shfl_down(s, off, 32);
    if (lane == 0) ebuf[l] = (mask[(size_t)b * L + l] > 0.5f) ? NEGV : s;
  }
  __syncthreads();

  float mx = -3.4e38f;
  for (int l = tid; l < L; l += 256) mx = fmaxf(mx, ebuf[l]);
  for (int off = 16; off > 0; off >>= 1) mx = fmaxf(mx, __shfl_down(mx, off, 32));
  if (lane == 0) red[wid] = mx;
  __syncthreads();
  if (tid == 0) {
    float v = red[0];
    for (int i = 1; i < 8; ++i) v = fmaxf(v, red[i]);
    red[0] = v;
  }
  __syncthreads();
  mx = red[0];
  __syncthreads();

  float sum = 0.f;
  for (int l = tid; l < L; l += 256) {
    float ex = __expf(ebuf[l] - mx);
    ebuf[l] = ex;
    sum += ex;
  }
  for (int off = 16; off > 0; off >>= 1) sum += __shfl_down(sum, off, 32);
  if (lane == 0) red[wid] = sum;
  __syncthreads();
  if (tid == 0) {
    float s = 0.f;
    for (int i = 0; i < 8; ++i) s += red[i];
    red[0] = s;
  }
  __syncthreads();
  const float inv = 1.f / red[0];
  for (int l = tid; l < L; l += 256) {
    float a = ebuf[l] * inv;
    ebuf[l] = a;
    a_out[(size_t)b * L + l] = a;
  }
  __syncthreads();

  for (int d = tid; d < DD; d += 256) {
    float acc = 0.f;
    for (int l = 0; l < L; ++l) acc += ebuf[l] * ctx[((size_t)l * BB + b) * DD + d];
    c_out[(size_t)b * DD + d] = acc;
  }
}

// mix = g*c + (1-g)*tc  (writes f32 + bf16 shadow, optional mix_final output)
__global__ void mix_kernel(const float* __restrict__ c, const float* __restrict__ tc,
                           const float* __restrict__ gate_f,
                           float* __restrict__ mix_f, unsigned short* __restrict__ mix_bf,
                           float* __restrict__ mixfinal)
{
  int i = blockIdx.x * 256 + threadIdx.x;
  if (i >= BB * DD) return;
  int b = i >> 9;
  float g = gate_f[b];
  float mval = g * c[i] + (1.f - g) * tc[i];
  mix_f[i]  = mval;
  mix_bf[i] = f2bf(mval);
  if (mixfinal) mixfinal[i] = mval;
}

// ---------------------------------------------------------------------------
// Readout GEMM [B,3D]x[3D,D]^T + bias, fused maxout(POOL=2) via lane shuffle.
// grid = (D/16, B/16), block = 32.
// ---------------------------------------------------------------------------
__global__ void readout_kernel(const unsigned short* __restrict__ Aemb,
                               const unsigned short* __restrict__ Aout,
                               const unsigned short* __restrict__ Amix,
                               const unsigned short* __restrict__ W,
                               const float* __restrict__ bias,
                               float* __restrict__ gout)
{
  const int col0 = blockIdx.x * 16;
  const int row0 = blockIdx.y * 16;
  const int lane = threadIdx.x & 31;
  const int m  = lane & 15;
  const int hi = lane >> 4;
  const unsigned short* Brow = W + (size_t)(col0 + m) * (3 * DD);
  v8f acc = {};
  for (int k0 = 0; k0 < 3 * DD; k0 += 32) {
    const int seg = k0 >> 9;
    const unsigned short* Aseg = (seg == 0) ? Aemb : ((seg == 1) ? Aout : Amix);
    const unsigned short* Arow = Aseg + (size_t)(row0 + m) * DD + (k0 & (DD - 1));
    FragBF a, b;
    a.u[0] = *(const u32x4*)(Arow + hi * 8);
    a.u[1] = *(const u32x4*)(Arow + 16 + hi * 8);
    b.u[0] = *(const u32x4*)(Brow + k0 + hi * 8);
    b.u[1] = *(const u32x4*)(Brow + k0 + 16 + hi * 8);
    acc = __builtin_amdgcn_wmma_f32_16x16x32_bf16(false, a.v, false, b.v,
                                                  (short)0, acc, false, false);
  }
  const int col = col0 + m;
  const float bv = bias[col];
#pragma unroll
  for (int r = 0; r < 8; ++r) {
    float v0 = acc[r] + bv;
    float v1 = __shfl_xor(v0, 1, 32);     // partner column (col ^ 1), incl. its bias
    if ((m & 1) == 0) {
      int row = row0 + hi * 8 + r;
      gout[(size_t)row * (DD / 2) + (col >> 1)] = fmaxf(v0, v1);
    }
  }
}

// -------------------------- small helper kernels ---------------------------
__global__ void f2bf_kernel(const float* __restrict__ src,
                            unsigned short* __restrict__ dst, int n) {
  int i = blockIdx.x * 256 + threadIdx.x;
  if (i < n) dst[i] = f2bf(src[i]);
}

__global__ void initstate_kernel(const float* __restrict__ src, float* __restrict__ dstf,
                                 unsigned short* __restrict__ dstb, int n) {
  int i = blockIdx.x * 256 + threadIdx.x;
  if (i < n) { float v = src[i]; dstf[i] = v; dstb[i] = f2bf(v); }
}

__global__ void emb_gather_kernel(const int* __restrict__ tgt,
                                  const float* __restrict__ table,
                                  unsigned short* __restrict__ dst, int ntokens) {
  int i = blockIdx.x * 256 + threadIdx.x;
  if (i < ntokens * DD) {
    int tokidx = i >> 9;
    int d = i & (DD - 1);
    dst[i] = f2bf(table[(size_t)tgt[tokidx] * DD + d]);
  }
}

__global__ void hfinal_kernel(const float* __restrict__ h0, const float* __restrict__ h1,
                              float* __restrict__ dst) {
  int i = blockIdx.x * 256 + threadIdx.x;
  if (i < BB * DD) { dst[i] = h0[i]; dst[BB * DD + i] = h1[i]; }
}

// ---------------------------------------------------------------------------
extern "C" void kernel_launch(void* const* d_in, const int* in_sizes, int n_in,
                              void* d_out, int out_size, void* d_ws, size_t ws_size,
                              hipStream_t stream)
{
  (void)in_sizes; (void)n_in; (void)out_size; (void)ws_size;
  const int*   tgt       = (const int*)  d_in[0];
  const float* hidden    = (const float*)d_in[1];
  const float* context   = (const float*)d_in[2];
  const float* smask     = (const float*)d_in[3];
  const float* tcontext  = (const float*)d_in[4];
  const float* tmask     = (const float*)d_in[5];
  const float* mixinit   = (const float*)d_in[6];
  const float* emb_table = (const float*)d_in[7];
  const float* Wih0      = (const float*)d_in[8];
  const float* Whh0      = (const float*)d_in[9];
  const float* bih0      = (const float*)d_in[10];
  const float* bhh0      = (const float*)d_in[11];
  const float* Wih1      = (const float*)d_in[12];
  const float* Whh1      = (const float*)d_in[13];
  const float* bih1      = (const float*)d_in[14];
  const float* bhh1      = (const float*)d_in[15];
  const float* preW      = (const float*)d_in[16];
  const float* preB      = (const float*)d_in[17];
  const float* qW        = (const float*)d_in[18];
  const float* vA        = (const float*)d_in[19];
  const float* tpreW     = (const float*)d_in[20];
  const float* tpreB     = (const float*)d_in[21];
  const float* tqW       = (const float*)d_in[22];
  const float* tvA       = (const float*)d_in[23];
  const float* roW       = (const float*)d_in[24];
  const float* roB       = (const float*)d_in[25];
  const float* gW        = (const float*)d_in[26];
  const float* gB        = (const float*)d_in[27];

  float* out = (float*)d_out;
  const size_t GOUT_OFF  = 0;
  const size_t HFIN_OFF  = (size_t)TT * BB * (DD / 2);
  const size_t ATTN_OFF  = HFIN_OFF + (size_t)2 * BB * DD;
  const size_t TATTN_OFF = ATTN_OFF + (size_t)TT * BB * SS;
  const size_t MIXF_OFF  = TATTN_OFF + (size_t)TT * BB * KK;
  const size_t GATE_OFF  = MIXF_OFF + (size_t)BB * DD;

  char* wp_ = (char*)d_ws;
  auto take = [&](size_t bytes) -> void* {
    void* r = (void*)wp_;
    wp_ += (bytes + 255) & ~(size_t)255;
    return r;
  };

  unsigned short* wih0_bf  = (unsigned short*)take((size_t)3 * DD * 2 * DD * 2);
  unsigned short* whh0_bf  = (unsigned short*)take((size_t)3 * DD * DD * 2);
  unsigned short* wih1_bf  = (unsigned short*)take((size_t)3 * DD * DD * 2);
  unsigned short* whh1_bf  = (unsigned short*)take((size_t)3 * DD * DD * 2);
  unsigned short* preW_bf  = (unsigned short*)take((size_t)DD * DD * 2);
  unsigned short* qW_bf    = (unsigned short*)take((size_t)DD * DD * 2);
  unsigned short* tpreW_bf = (unsigned short*)take((size_t)DD * DD * 2);
  unsigned short* tqW_bf   = (unsigned short*)take((size_t)DD * DD * 2);
  unsigned short* roW_bf   = (unsigned short*)take((size_t)DD * 3 * DD * 2);
  unsigned short* emb_bf   = (unsigned short*)take((size_t)TT * BB * DD * 2);
  float* pre_f  = (float*)take((size_t)BB * SS * DD * 4);
  float* tpre_f = (float*)take((size_t)BB * KK * DD * 4);
  float* h0f[2]; h0f[0] = (float*)take((size_t)BB * DD * 4); h0f[1] = (float*)take((size_t)BB * DD * 4);
  float* h1f[2]; h1f[0] = (float*)take((size_t)BB * DD * 4); h1f[1] = (float*)take((size_t)BB * DD * 4);
  unsigned short* h0b[2]; h0b[0] = (unsigned short*)take((size_t)BB * DD * 2); h0b[1] = (unsigned short*)take((size_t)BB * DD * 2);
  unsigned short* h1b[2]; h1b[0] = (unsigned short*)take((size_t)BB * DD * 2); h1b[1] = (unsigned short*)take((size_t)BB * DD * 2);
  float* mix_f  = (float*)take((size_t)BB * DD * 4);
  unsigned short* mix_bf = (unsigned short*)take((size_t)BB * DD * 2);
  float* q_f    = (float*)take((size_t)BB * DD * 4);
  float* tq_f   = (float*)take((size_t)BB * DD * 4);
  float* c_f    = (float*)take((size_t)BB * DD * 4);
  float* tc_f   = (float*)take((size_t)BB * DD * 4);
  float* gate_f = (float*)take((size_t)BB * 4);

  auto cvt = [&](const float* s, unsigned short* d, int n) {
    f2bf_kernel<<<(n + 255) / 256, 256, 0, stream>>>(s, d, n);
  };
  cvt(Wih0, wih0_bf, 3 * DD * 2 * DD);
  cvt(Whh0, whh0_bf, 3 * DD * DD);
  cvt(Wih1, wih1_bf, 3 * DD * DD);
  cvt(Whh1, whh1_bf, 3 * DD * DD);
  cvt(preW, preW_bf, DD * DD);
  cvt(qW,   qW_bf,   DD * DD);
  cvt(tpreW, tpreW_bf, DD * DD);
  cvt(tqW,  tqW_bf,  DD * DD);
  cvt(roW,  roW_bf,  DD * 3 * DD);

  emb_gather_kernel<<<(TT * BB * DD + 255) / 256, 256, 0, stream>>>(tgt, emb_table, emb_bf, TT * BB);
  initstate_kernel<<<(BB * DD + 255) / 256, 256, 0, stream>>>(hidden,            h0f[0], h0b[0], BB * DD);
  initstate_kernel<<<(BB * DD + 255) / 256, 256, 0, stream>>>(hidden + BB * DD,  h1f[0], h1b[0], BB * DD);
  initstate_kernel<<<(BB * DD + 255) / 256, 256, 0, stream>>>(mixinit,           mix_f,  mix_bf, BB * DD);

  gemm_pre_kernel<<<dim3(DD / 16, BB * SS / 16), 32, 0, stream>>>(context,  SS, preW_bf,  preB,  pre_f);
  gemm_pre_kernel<<<dim3(DD / 16, BB * KK / 16), 32, 0, stream>>>(tcontext, KK, tpreW_bf, tpreB, tpre_f);

  for (int t = 0; t < TT; ++t) {
    const int rp = t & 1, wpar = rp ^ 1;
    const unsigned short* emb_t = emb_bf + (size_t)t * BB * DD;

    gru_step_kernel<<<dim3(DD / 16, BB / 16), 32, 0, stream>>>(
        emb_t, mix_bf, 2 * DD, wih0_bf, whh0_bf,
        h0b[rp], h0f[rp], bih0, bhh0, h0f[wpar], h0b[wpar]);
    gru_step_kernel<<<dim3(DD / 16, BB / 16), 32, 0, stream>>>(
        h0b[wpar], (const unsigned short*)nullptr, DD, wih1_bf, whh1_bf,
        h1b[rp], h1f[rp], bih1, bhh1, h1f[wpar], h1b[wpar]);

    gate_kernel<<<BB, 32, 0, stream>>>(h1f[wpar], gW, gB, gate_f,
                                       out + GATE_OFF + (size_t)t * BB);

    gemm_bf16_kernel<<<dim3(DD / 16, BB / 16), 32, 0, stream>>>(
        h1b[wpar], DD, qW_bf, DD, q_f, DD, DD, (const float*)nullptr);
    gemm_bf16_kernel<<<dim3(DD / 16, BB / 16), 32, 0, stream>>>(
        h1b[wpar], DD, tqW_bf, DD, tq_f, DD, DD, (const float*)nullptr);

    attention_kernel<<<BB, 256, 0, stream>>>(pre_f, q_f, vA, context, smask, SS,
        c_f,  out + ATTN_OFF  + (size_t)t * BB * SS);
    attention_kernel<<<BB, 256, 0, stream>>>(tpre_f, tq_f, tvA, tcontext, tmask, KK,
        tc_f, out + TATTN_OFF + (size_t)t * BB * KK);

    mix_kernel<<<(BB * DD + 255) / 256, 256, 0, stream>>>(
        c_f, tc_f, gate_f, mix_f, mix_bf,
        (t == TT - 1) ? (out + MIXF_OFF) : (float*)nullptr);

    readout_kernel<<<dim3(DD / 16, BB / 16), 32, 0, stream>>>(
        emb_t, h1b[wpar], mix_bf, roW_bf, roB,
        out + GOUT_OFF + (size_t)t * BB * (DD / 2));
  }

  hfinal_kernel<<<(BB * DD + 255) / 256, 256, 0, stream>>>(h0f[TT & 1], h1f[TT & 1],
                                                           out + HFIN_OFF);
}
